// LSTMModel_29472065585350
// MI455X (gfx1250) — matrix operand order repaired and checked by reading, not compiled, not run on hardware
//
#include <hip/hip_runtime.h>

typedef __attribute__((ext_vector_type(16))) _Float16 v16h;
typedef __attribute__((ext_vector_type(8)))  _Float16 v8h;
typedef __attribute__((ext_vector_type(8)))  float    v8f;

#define BT    16          // batch tile rows (WMMA M)
#define DIN   64          // input feature dim D
#define HD    128         // hidden dim H
#define G4    512         // 4*H gate width
#define TPB   512         // 16 waves of 32
#define TLEN  1024
#define BATCH 512

#define LOG2E 1.4426950408889634f

// Branch-free fast activations on raw v_exp_f32 (exp2) + v_rcp_f32.
__device__ __forceinline__ float fast_sigmoid(float x) {
    return __builtin_amdgcn_rcpf(1.f + __builtin_amdgcn_exp2f(-LOG2E * x));
}
__device__ __forceinline__ float fast_tanh(float x) {
    return 1.f - 2.f * __builtin_amdgcn_rcpf(1.f + __builtin_amdgcn_exp2f(2.f * LOG2E * x));
}

// ---- A-fragment (16x32 f16, MxK) from row-major f16 LDS tile -------------
__device__ __forceinline__ v16h a_frag(const _Float16* __restrict__ s,
                                       int ldk, int kc, int lane) {
    int row = lane & 15;
    int kb  = kc * 32 + ((lane & 16) ? 8 : 0);
    v8h lo = *(const v8h*)(s + row * ldk + kb);        // 16B ds_load
    v8h hi = *(const v8h*)(s + row * ldk + kb + 16);   // 16B ds_load
    v16h r;
#pragma unroll
    for (int e = 0; e < 8; ++e) { r[e] = lo[e]; r[e + 8] = hi[e]; }
    return r;
}

// ---- B-fragment (32x16 f16, KxN) built once from global f32 weights ------
__device__ __forceinline__ v16h w_frag(const float* __restrict__ W,
                                       int ncol, int ldk, int kc, int lane) {
    int base = ncol * ldk + kc * 32 + ((lane & 16) ? 8 : 0);
    v16h r;
#pragma unroll
    for (int e = 0; e < 16; ++e) r[e] = (_Float16)W[base + e + (e >= 8 ? 8 : 0)];
    return r;
}

__global__ __launch_bounds__(TPB)
void lstm_fused_kernel(const float* __restrict__ x,
                       const float* __restrict__ Wih,
                       const float* __restrict__ Whh,
                       const float* __restrict__ bih,
                       const float* __restrict__ bhh,
                       const float* __restrict__ W1,
                       const float* __restrict__ b1,
                       const float* __restrict__ W2,
                       const float* __restrict__ b2,
                       float* __restrict__ out) {
    __shared__ _Float16 x16[BT * DIN];   // staged x_t tile (f16)
    __shared__ _Float16 h16[BT * HD];    // h in f16 for WMMA A operand
    __shared__ float    hf [BT * HD];    // final-step h (f32) for the head
    __shared__ float    gates[BT * G4];  // f32 gate pre-activations (bias folded)
    __shared__ float    zbuf[BT * 64];   // head intermediates

    const int tid  = threadIdx.x;
    const int lane = tid & 31;
    const int wv   = tid >> 5;                       // 0..15
    const int bb   = blockIdx.x;                     // batch tile id
    const float* xblk = x + (size_t)bb * BT * TLEN * DIN;

    // ---- one-time: recurrent + input weights into VGPRs as f16 B-frags ---
    v16h bwih[2][2];   // two N-tiles per wave, K=64 -> 2 chunks
    v16h bwhh[2][4];   // K=128 -> 4 chunks
    float bc[2];       // per-lane gate-column bias (same for all 8 C rows)
#pragma unroll
    for (int j = 0; j < 2; ++j) {
        int ncol = (wv * 2 + j) * 16 + (lane & 15);  // row of W (= gate col)
        bc[j] = bih[ncol] + bhh[ncol];
#pragma unroll
        for (int kc = 0; kc < 2; ++kc) bwih[j][kc] = w_frag(Wih, ncol, DIN, kc, lane);
#pragma unroll
        for (int kc = 0; kc < 4; ++kc) bwhh[j][kc] = w_frag(Whh, ncol, HD, kc, lane);
    }

    // ---- per-thread elementwise assignment: 4 rows x 1 hidden column -----
    const int hh = tid & 127;
    const int r0 = (tid >> 7) * 4;
    float creg[4] = {0.f, 0.f, 0.f, 0.f};
    float hreg[4] = {0.f, 0.f, 0.f, 0.f};

    // staging indices for the x tile (2 elements / thread)
    const int i0 = tid, i1 = tid + TPB;
    const size_t xoff0 = (size_t)(i0 >> 6) * TLEN * DIN + (i0 & 63);
    const size_t xoff1 = (size_t)(i1 >> 6) * TLEN * DIN + (i1 & 63);

    for (int i = tid; i < BT * HD; i += TPB) h16[i] = (_Float16)0.f;

    // pipeline prologue: commit tile 0 to LDS, stage tile 1 in registers
    x16[i0] = (_Float16)xblk[xoff0];
    x16[i1] = (_Float16)xblk[xoff1];
    const float* x1 = xblk + DIN;
    float xr0 = x1[xoff0];
    float xr1 = x1[xoff1];
    __syncthreads();

    // =============== recurrent time loop: 2 barriers / step ===============
    for (int t = 0; t < TLEN; ++t) {
        // ---- GEMM phase: gates = bias + x_t*Wih^T + h*Whh^T --------------
        v16h ax0 = a_frag(x16, DIN, 0, lane);
        v16h ax1 = a_frag(x16, DIN, 1, lane);
        v16h ah0 = a_frag(h16, HD, 0, lane);
        v16h ah1 = a_frag(h16, HD, 1, lane);
        v16h ah2 = a_frag(h16, HD, 2, lane);
        v16h ah3 = a_frag(h16, HD, 3, lane);

#pragma unroll
        for (int j = 0; j < 2; ++j) {
            v8f acc;               // bias pre-folded into the accumulator
#pragma unroll
            for (int v = 0; v < 8; ++v) acc[v] = bc[j];
            acc = __builtin_amdgcn_wmma_f32_16x16x32_f16(false, ax0, false, bwih[j][0], (short)0, acc, false, false);
            acc = __builtin_amdgcn_wmma_f32_16x16x32_f16(false, ax1, false, bwih[j][1], (short)0, acc, false, false);
            acc = __builtin_amdgcn_wmma_f32_16x16x32_f16(false, ah0, false, bwhh[j][0], (short)0, acc, false, false);
            acc = __builtin_amdgcn_wmma_f32_16x16x32_f16(false, ah1, false, bwhh[j][1], (short)0, acc, false, false);
            acc = __builtin_amdgcn_wmma_f32_16x16x32_f16(false, ah2, false, bwhh[j][2], (short)0, acc, false, false);
            acc = __builtin_amdgcn_wmma_f32_16x16x32_f16(false, ah3, false, bwhh[j][3], (short)0, acc, false, false);
            // scatter C/D tile: VGPR v -> row v (+8 for upper lane half)
            int col   = (wv * 2 + j) * 16 + (lane & 15);
            int rbase = (lane & 16) ? 8 : 0;
#pragma unroll
            for (int v = 0; v < 8; ++v) gates[(rbase + v) * G4 + col] = acc[v];
        }
        __syncthreads();   // gates ready; all x16/h16 reads of step t done

        // ---- elementwise phase (also hides next-x staging) ---------------
        // commit register-staged tile t+1; issue loads for tile t+2
        if (t + 1 < TLEN) {
            x16[i0] = (_Float16)xr0;
            x16[i1] = (_Float16)xr1;
        }
        if (t + 2 < TLEN) {
            const float* xnxt = xblk + (size_t)(t + 2) * DIN;
            xr0 = xnxt[xoff0];
            xr1 = xnxt[xoff1];
        }
        if (t + 6 < TLEN) {   // long-range prefetch: HBM -> cache
            __builtin_prefetch(xblk + (size_t)(t + 6) * DIN + xoff0, 0, 3);
        }

        // LSTM cell update (f32), gate order i,f,g,o — fully branch-free
#pragma unroll
        for (int k = 0; k < 4; ++k) {
            int r = r0 + k;
            float si = fast_sigmoid(gates[r * G4 + 0 * HD + hh]);
            float sf = fast_sigmoid(gates[r * G4 + 1 * HD + hh]);
            float tg = fast_tanh   (gates[r * G4 + 2 * HD + hh]);
            float so = fast_sigmoid(gates[r * G4 + 3 * HD + hh]);
            float c  = sf * creg[k] + si * tg;
            creg[k]  = c;
            float h  = so * fast_tanh(c);
            hreg[k]  = h;
            h16[r * HD + hh] = (_Float16)h;   // only f16 h feeds the GEMM
        }
        __syncthreads();   // h16 (and next x16) ready for step t+1 GEMM
    }

    // spill final-step h (f32) once for the FC head
#pragma unroll
    for (int k = 0; k < 4; ++k) hf[(r0 + k) * HD + hh] = hreg[k];
    __syncthreads();

    // ===================== FC head: relu(h W1^T + b1) W2^T + b2 ===========
    if (tid < BT * 64) {
        int r = tid >> 6, j = tid & 63;
        float s = b1[j];
        const float* w1r = W1 + j * HD;
        const float* hr  = hf + r * HD;
#pragma unroll 8
        for (int k = 0; k < HD; ++k) s += hr[k] * w1r[k];
        zbuf[tid] = fmaxf(s, 0.f) * W2[j];
    }
    __syncthreads();
    if (tid < BT) {
        float y = b2[0];
#pragma unroll
        for (int j = 0; j < 64; ++j) y += zbuf[tid * 64 + j];
        out[bb * BT + tid] = y;
    }
}

extern "C" void kernel_launch(void* const* d_in, const int* in_sizes, int n_in,
                              void* d_out, int out_size, void* d_ws, size_t ws_size,
                              hipStream_t stream) {
    (void)in_sizes; (void)n_in; (void)d_ws; (void)ws_size; (void)out_size;
    const float* x   = (const float*)d_in[0];
    const float* Wih = (const float*)d_in[1];
    const float* Whh = (const float*)d_in[2];
    const float* bih = (const float*)d_in[3];
    const float* bhh = (const float*)d_in[4];
    const float* W1  = (const float*)d_in[5];
    const float* b1  = (const float*)d_in[6];
    const float* W2  = (const float*)d_in[7];
    const float* b2  = (const float*)d_in[8];
    float* out = (float*)d_out;

    dim3 grid(BATCH / BT);   // 32 independent batch tiles, no cross-WG sync
    dim3 block(TPB);         // 16 wave32s per workgroup
    lstm_fused_kernel<<<grid, block, 0, stream>>>(x, Wih, Whh, bih, bhh,
                                                  W1, b1, W2, b2, out);
}
